// CDEM_62079457296798
// MI455X (gfx1250) — matrix operand order repaired and checked by reading, not compiled, not run on hardware
//
#include <hip/hip_runtime.h>
#include <hip/hip_bf16.h>
#include <math.h>

#define HW   16384
#define IMGW 128
#define BATCH 4

typedef _Float16 v16h __attribute__((ext_vector_type(16)));
typedef _Float16 v8h  __attribute__((ext_vector_type(8)));
typedef float    v8f  __attribute__((ext_vector_type(8)));

static __device__ __forceinline__ v8f wmma_f16(v16h a, v16h b, v8f c) {
  // D = A(16x32 f16) * B(32x16 f16) + C(16x16 f32)
  return __builtin_amdgcn_wmma_f32_16x16x32_f16(false, a, false, b, (short)0, c, false, false);
}

union AFrag { v16h v; v8h h2[2]; };

// CDNA5 async global->LDS copy (16B per lane), tracked by ASYNCcnt.
// vdst operand = per-lane LDS byte address (LDS aperture: addr[31:0]).
static __device__ __forceinline__ void async_copy_b128(unsigned lds_byte_off,
                                                       const void* gptr) {
  asm volatile("global_load_async_to_lds_b128 %0, %1, off"
               :: "v"(lds_byte_off), "v"((unsigned long long)gptr)
               : "memory");
}
static __device__ __forceinline__ void async_wait0() {
  asm volatile("s_wait_asynccnt 0x0" ::: "memory");
}

// Stage one 32-K slab: weights via async-to-LDS, input via transposed stores.
static __device__ __forceinline__ void stage_tiles(
    _Float16* Wd, _Float16* Id,
    const _Float16* __restrict__ Wb, const _Float16* __restrict__ inB,
    int o0, int p0, int k0, int Cin, int t)
{
  // weights: 64 rows x 32 k, one async 16B copy per thread
  async_copy_b128((unsigned)(unsigned long long)(const void*)&Wd[t * 8],
                  Wb + (long)(o0 + (t >> 2)) * Cin + k0 + (t & 3) * 8);
  // input transposed to [p][k]: b32 loads (2 px), b16 scatter stores
  const int p2 = t & 31, c0 = t >> 5;
  #pragma unroll
  for (int pass = 0; pass < 4; ++pass) {
    const int c = c0 + pass * 8;
    const _Float16* src = inB + (long)(k0 + c) * HW + p0 + p2 * 2;
    const _Float16 v0 = src[0], v1 = src[1];
    Id[(p2 * 2 + 0) * 32 + c] = v0;
    Id[(p2 * 2 + 1) * 32 + c] = v1;
  }
}

// ---------------------------------------------------------------------------
// 1x1-conv GEMM, f16 in / f32-accumulate: out[b][o][p] = epi(sum_c W[o][c] in[c][p])
// Block 256 thr (8 waves). Tile: 64 out-ch x 64 pixels -> 16 wmma per K-step.
// Double-buffered: async weight copy + input staging for step i+1 overlap the
// WMMAs of step i; one barrier per K-step.
// EPI: 0=none 1=gelu 2: s1*v + s0*add  3: s0*add + s1*v   (add is f16)
// ---------------------------------------------------------------------------
template <int EPI, bool OUTF16>
__global__ __launch_bounds__(256) void k_gemm1x1(
    const _Float16* __restrict__ inp, long inBatchStride,
    const _Float16* __restrict__ W, long wBatchStride, int Cin, int Cout,
    void* __restrict__ outp,
    const _Float16* __restrict__ addp,
    const float* __restrict__ s0p, const float* __restrict__ s1p)
{
  __shared__ _Float16 Wl[2 * 64 * 32];   // [buf][o_local][k]  k-contig, 64B rows
  __shared__ _Float16 Il[2 * 64 * 32];   // [buf][p_local][k]  k-contig, 64B rows

  const int t    = threadIdx.x;
  const int lane = t & 31;
  const int wv   = t >> 5;
  const int wo   = wv >> 1;          // 0..3  out-ch subtile
  const int wp   = wv & 1;           // 0..1  pixel half (32 px each)
  const int p0   = blockIdx.x * 64;
  const int o0   = blockIdx.y * 64;
  const int b    = blockIdx.z;

  const _Float16* inB = inp + (long)b * inBatchStride;
  const _Float16* Wb  = W + (long)b * wBatchStride;

  v8f acc0 = {}, acc1 = {};

  // prologue: fill buffer 0
  stage_tiles(&Wl[0], &Il[0], Wb, inB, o0, p0, 0, Cin, t);
  async_wait0();
  __syncthreads();

  int buf = 0;
  for (int k0 = 0; k0 < Cin; k0 += 32) {
    const bool more = (k0 + 32 < Cin);
    if (more)
      stage_tiles(&Wl[(buf ^ 1) * 2048], &Il[(buf ^ 1) * 2048],
                  Wb, inB, o0, p0, k0 + 32, Cin, t);

    // A fragment (16x32): row = lane&15, two 8-half chunks
    const _Float16* Arow = &Wl[buf * 2048 + (wo * 16 + (lane & 15)) * 32];
    const int ab = (lane & 16) ? 16 : 0;
    AFrag a;
    a.h2[0] = *(const v8h*)(Arow + ab);
    a.h2[1] = *(const v8h*)(Arow + ab + 8);
    // two B fragments (32x16): col = lane&15, 16 contiguous k halves
    const int bo = (lane & 16) ? 16 : 0;
    const _Float16* B0 = &Il[buf * 2048 + (wp * 32 + 0  + (lane & 15)) * 32 + bo];
    const _Float16* B1 = &Il[buf * 2048 + (wp * 32 + 16 + (lane & 15)) * 32 + bo];
    v16h b0 = *(const v16h*)B0;
    v16h b1 = *(const v16h*)B1;

    acc0 = wmma_f16(a.v, b0, acc0);
    acc1 = wmma_f16(a.v, b1, acc1);

    if (more) {
      async_wait0();     // this wave's async copies into buf^1 done
      __syncthreads();   // publish buf^1; retire all reads of buf
      buf ^= 1;
    }
  }

  const float s0 = (EPI == 2 || EPI == 3) ? *s0p : 0.f;
  const float s1 = (EPI == 2 || EPI == 3) ? *s1p : 0.f;
  const int obase = o0 + wo * 16 + ((lane & 16) ? 8 : 0);

  float*          outF = (float*)outp    + (long)b * Cout * HW;
  _Float16*       outH = (_Float16*)outp + (long)b * Cout * HW;
  const _Float16* addH = addp            + (long)b * Cout * HW;

  #pragma unroll
  for (int nt = 0; nt < 2; ++nt) {
    const int pg = p0 + wp * 32 + nt * 16 + (lane & 15);
    #pragma unroll
    for (int r = 0; r < 8; ++r) {
      const long idx = (long)(obase + r) * HW + pg;
      float v = (nt == 0) ? acc0[r] : acc1[r];
      if (EPI == 1) {
        v = 0.5f * v * (1.0f + erff(v * 0.70710678118654752f));   // exact GELU
      } else if (EPI == 2) {
        v = s1 * v + s0 * (float)addH[idx];
      } else if (EPI == 3) {
        v = s0 * (float)addH[idx] + s1 * v;
      }
      if (OUTF16) outH[idx] = (_Float16)v; else outF[idx] = v;
    }
  }
}

// ---------------------------------------------------------------------------
// G[bh][c][d] += sum_{p in segment} q[b,h*48+c,p] * k[b,h*48+d,p]
// grid (16, 4): 4 pixel segments, atomicAdd f32 combine. 9 waves/block, one
// 16x16 tile each, 4 independent accumulator chains.
// ---------------------------------------------------------------------------
__global__ __launch_bounds__(288) void k_qkT(const _Float16* __restrict__ q,
                                             const _Float16* __restrict__ kk,
                                             float* __restrict__ G)
{
  const int t = threadIdx.x, lane = t & 31, wv = t >> 5;
  const int mt = wv / 3, nt = wv % 3;
  const int bh = blockIdx.x, b = bh >> 2, h = bh & 3;
  const int kbeg = blockIdx.y * (HW / 4), kend = kbeg + (HW / 4);

  const _Float16* qr = q  + ((long)b * 192 + h * 48 + mt * 16 + (lane & 15)) * HW;
  const _Float16* kr = kk + ((long)b * 384 + h * 48 + nt * 16 + (lane & 15)) * HW;
  const int ab  = (lane & 16) ? 16 : 0;
  const int bb0 = (lane & 16) ? 16 : 0;

  v8f acc[4] = {{}, {}, {}, {}};
  for (int kp = kbeg; kp < kend; kp += 128) {
    #pragma unroll
    for (int u = 0; u < 4; ++u) {
      const int ko = kp + u * 32;
      AFrag a;
      a.h2[0] = *(const v8h*)(qr + ko + ab);
      a.h2[1] = *(const v8h*)(qr + ko + ab + 8);
      v16h bf = *(const v16h*)(kr + ko + bb0);
      acc[u] = wmma_f16(a.v, bf, acc[u]);
    }
  }
  v8f s = acc[0] + acc[1] + acc[2] + acc[3];
  #pragma unroll
  for (int r = 0; r < 8; ++r) {
    const int m = mt * 16 + r + ((lane & 16) ? 8 : 0);
    const int n = nt * 16 + (lane & 15);
    atomicAdd(&G[(long)bh * 2304 + m * 48 + n], s[r]);
  }
}

// Row L2-norms of q (192ch layout) and k (first 192 of 384ch layout).
__global__ __launch_bounds__(256) void k_norm(const _Float16* __restrict__ q,
                                              const _Float16* __restrict__ kk,
                                              float* __restrict__ nq,
                                              float* __restrict__ nk)
{
  __shared__ float red[256];
  const int row = blockIdx.x;        // b*192 + c
  const int which = blockIdx.y;      // 0=q, 1=k
  const _Float16* ptr;
  if (which == 0) ptr = q + (long)row * HW;
  else { int b = row / 192, c = row % 192; ptr = kk + ((long)b * 384 + c) * HW; }
  float s = 0.f;
  for (int i = threadIdx.x * 8; i < HW; i += 256 * 8) {
    v8h v = *(const v8h*)(ptr + i);
    #pragma unroll
    for (int j = 0; j < 8; ++j) { float f = (float)v[j]; s += f * f; }
  }
  red[threadIdx.x] = s; __syncthreads();
  for (int off = 128; off > 0; off >>= 1) {
    if (threadIdx.x < off) red[threadIdx.x] += red[threadIdx.x + off];
    __syncthreads();
  }
  if (threadIdx.x == 0) (which == 0 ? nq : nk)[row] = fmaxf(sqrtf(red[0]), 1e-12f);
}

// softmax((G / (||q_c|| ||k_d||)) * temp[h]) -> block-diagonal 192x192 f16
// per-batch mixing weight (attn@v becomes a per-batch 1x1 conv).
__global__ __launch_bounds__(64) void k_softmax_wz(const float* __restrict__ G,
                                                   const float* __restrict__ nq,
                                                   const float* __restrict__ nk,
                                                   const float* __restrict__ temp,
                                                   _Float16* __restrict__ Wz)
{
  const int bh = blockIdx.x, b = bh >> 2, h = bh & 3;
  const int c = threadIdx.x;
  if (c >= 48) return;
  const float tp = temp[h];
  const float invq = 1.0f / nq[b * 192 + h * 48 + c];
  float l[48];
  float m = -1e30f;
  for (int d = 0; d < 48; ++d) {
    float v = G[(long)bh * 2304 + c * 48 + d] * invq / nk[b * 192 + h * 48 + d] * tp;
    l[d] = v; m = fmaxf(m, v);
  }
  float s = 0.f;
  for (int d = 0; d < 48; ++d) { l[d] = expf(l[d] - m); s += l[d]; }
  const float inv = 1.0f / s;
  _Float16* row = Wz + ((long)b * 192 + h * 48 + c) * 192;
  for (int col = 0; col < 192; ++col) row[col] = (_Float16)0.f;
  for (int d = 0; d < 48; ++d) row[h * 48 + d] = (_Float16)(l[d] * inv);
}

// depthwise 3x3, SAME zero padding, cross-correlation (JAX conv semantics)
__global__ __launch_bounds__(256) void k_dwconv(const _Float16* __restrict__ in,
                                                const float* __restrict__ w9,
                                                _Float16* __restrict__ out, int C)
{
  const long bc = blockIdx.y;
  const int c = (int)(bc % C);
  const int p = blockIdx.x * 256 + threadIdx.x;
  const int h = p >> 7, x = p & 127;
  const float* wc = w9 + c * 9;
  const _Float16* ib = in + bc * HW;
  float acc = 0.f;
  #pragma unroll
  for (int dy = 0; dy < 3; ++dy) {
    const int hh = h + dy - 1;
    if ((unsigned)hh < 128u) {
      #pragma unroll
      for (int dx = 0; dx < 3; ++dx) {
        const int xx = x + dx - 1;
        if ((unsigned)xx < 128u) acc += (float)ib[hh * IMGW + xx] * wc[dy * 3 + dx];
      }
    }
  }
  out[bc * HW + p] = (_Float16)acc;
}

// vectorized f32 -> f16 (n must be a multiple of 8)
__global__ __launch_bounds__(256) void k_f32h8(const float* __restrict__ s,
                                               _Float16* __restrict__ d, int n)
{
  const int i = (blockIdx.x * 256 + threadIdx.x) * 8;
  if (i < n) {
    const float4 a = *(const float4*)(s + i);
    const float4 c = *(const float4*)(s + i + 4);
    v8h o = { (_Float16)a.x, (_Float16)a.y, (_Float16)a.z, (_Float16)a.w,
              (_Float16)c.x, (_Float16)c.y, (_Float16)c.z, (_Float16)c.w };
    *(v8h*)(d + i) = o;
  }
}

__global__ __launch_bounds__(256) void k_zero(float* __restrict__ p, int n)
{
  const int i = blockIdx.x * 256 + threadIdx.x;
  if (i < n) p[i] = 0.f;
}

// ---------------------------------------------------------------------------
extern "C" void kernel_launch(void* const* d_in, const int* in_sizes, int n_in,
                              void* d_out, int out_size, void* d_ws, size_t ws_size,
                              hipStream_t stream) {
  const float* x        = (const float*)d_in[0];
  const float* y        = (const float*)d_in[1];
  const float* q_w      = (const float*)d_in[2];
  const float* q_dw_w   = (const float*)d_in[3];
  const float* kv_w     = (const float*)d_in[4];
  const float* kv_dw_w  = (const float*)d_in[5];
  const float* linear_w = (const float*)d_in[6];
  const float* proj_w   = (const float*)d_in[7];
  const float* ffn1_w   = (const float*)d_in[8];
  const float* ffn2_w   = (const float*)d_in[9];
  const float* temp     = (const float*)d_in[10];
  const float* alpha    = (const float*)d_in[11];
  const float* beta     = (const float*)d_in[12];
  const float* gamma    = (const float*)d_in[13];
  const float* delta    = (const float*)d_in[14];

  char* ws = (char*)d_ws;
  size_t off = 0;
  auto alloc = [&](size_t bytes) -> void* {
    void* p = ws + off;
    off = (off + bytes + 255) & ~((size_t)255);
    return p;
  };

  const size_t T192 = (size_t)BATCH * 192 * HW;   // elements
  _Float16* R1  = (_Float16*)alloc((size_t)BATCH * 768 * HW * 2); // kv_pre -> ffn_mid
  _Float16* R2  = (_Float16*)alloc((size_t)BATCH * 384 * HW * 2); // kv_dw (k|v)
  _Float16* R3  = (_Float16*)alloc(T192 * 2);                     // q_pre -> z -> t2
  _Float16* R4  = (_Float16*)alloc(T192 * 2);                     // q_dw -> t1
  _Float16* Rx  = (_Float16*)alloc(T192 * 2);                     // x (f16)
  _Float16* Ry  = (_Float16*)alloc(T192 * 2);                     // y (f16)
  _Float16* wq  = (_Float16*)alloc(192 * 192 * 2);
  _Float16* wkv = (_Float16*)alloc(384 * 192 * 2);
  _Float16* wln = (_Float16*)alloc(192 * 192 * 2);
  _Float16* wf1 = (_Float16*)alloc(768 * 192 * 2);
  _Float16* wf2 = (_Float16*)alloc(192 * 768 * 2);
  _Float16* wpj = (_Float16*)alloc(192 * 192 * 2);
  _Float16* Wz  = (_Float16*)alloc((size_t)BATCH * 192 * 192 * 2);
  float*    G   = (float*)alloc(16 * 48 * 48 * 4);
  float*    nq  = (float*)alloc(768 * 4);
  float*    nk  = (float*)alloc(768 * 4);
  if (off > ws_size) return;  // insufficient workspace

  // 0) inputs + weights -> f16
  k_f32h8<<<(int)(T192/8 + 255)/256, 256, 0, stream>>>(x, Rx, (int)T192);
  k_f32h8<<<(int)(T192/8 + 255)/256, 256, 0, stream>>>(y, Ry, (int)T192);
  k_f32h8<<<(192*192/8+255)/256, 256, 0, stream>>>(q_w,      wq,  192*192);
  k_f32h8<<<(384*192/8+255)/256, 256, 0, stream>>>(kv_w,     wkv, 384*192);
  k_f32h8<<<(192*192/8+255)/256, 256, 0, stream>>>(linear_w, wln, 192*192);
  k_f32h8<<<(768*192/8+255)/256, 256, 0, stream>>>(ffn1_w,   wf1, 768*192);
  k_f32h8<<<(192*768/8+255)/256, 256, 0, stream>>>(ffn2_w,   wf2, 192*768);
  k_f32h8<<<(192*192/8+255)/256, 256, 0, stream>>>(proj_w,   wpj, 192*192);
  k_zero<<<(16*2304+255)/256, 256, 0, stream>>>(G, 16*2304);

  const dim3 g192(HW/64, 3, BATCH), g384(HW/64, 6, BATCH), g768(HW/64, 12, BATCH);

  // 1) q_pre = q_w @ x ; kv_pre = kv_w @ y
  k_gemm1x1<0, true><<<g192, 256, 0, stream>>>(Rx, 192L*HW, wq,  0, 192, 192, R3, nullptr, nullptr, nullptr);
  k_gemm1x1<0, true><<<g384, 256, 0, stream>>>(Ry, 192L*HW, wkv, 0, 192, 384, R1, nullptr, nullptr, nullptr);

  // 2) depthwise 3x3
  k_dwconv<<<dim3(HW/256, BATCH*192), 256, 0, stream>>>(R3, q_dw_w,  R4, 192);
  k_dwconv<<<dim3(HW/256, BATCH*384), 256, 0, stream>>>(R1, kv_dw_w, R2, 384);

  // 3) row norms + Q K^T + softmax -> block-diag mixing weight
  k_norm<<<dim3(BATCH*192, 2), 256, 0, stream>>>(R4, R2, nq, nk);
  k_qkT<<<dim3(16, 4), 288, 0, stream>>>(R4, R2, G);
  k_softmax_wz<<<16, 64, 0, stream>>>(G, nq, nk, temp, Wz);

  // 4) z = attn @ v  (per-batch block-diag 1x1 conv over v = R2[:,192:])
  k_gemm1x1<0, true><<<g192, 256, 0, stream>>>(R2 + (size_t)192*HW, 384L*HW, Wz, 192L*192, 192, 192,
                                               R3, nullptr, nullptr, nullptr);

  // 5) t1 = beta * linear(z) + alpha * y
  k_gemm1x1<2, true><<<g192, 256, 0, stream>>>(R3, 192L*HW, wln, 0, 192, 192, R4, Ry, alpha, beta);

  // 6) ffn_mid = gelu(ffn1 @ t1)
  k_gemm1x1<1, true><<<g768, 256, 0, stream>>>(R4, 192L*HW, wf1, 0, 192, 768, R1, nullptr, nullptr, nullptr);

  // 7) t2 = gamma * t1 + delta * (ffn2 @ ffn_mid)
  k_gemm1x1<3, true><<<g192, 256, 0, stream>>>(R1, 768L*HW, wf2, 0, 768, 192, R3, R4, gamma, delta);

  // 8) out = proj @ t2   (f32)
  k_gemm1x1<0, false><<<g192, 256, 0, stream>>>(R3, 192L*HW, wpj, 0, 192, 192, d_out, nullptr, nullptr, nullptr);
}
